// MultiHeadTransformerSparseLayer_72378788873009
// MI455X (gfx1250) — matrix operand order repaired and checked by reading, not compiled, not run on hardware
//
#include <hip/hip_runtime.h>
#include <hip/hip_bf16.h>

// ---------------------------------------------------------------------------
// MI455X (gfx1250): all GEMMs via v_wmma_f32_16x16x32_bf16; attention K/V
// tiles staged block-wide into LDS with global_load_async_to_lds_b128
// (ASYNCcnt double-buffered) so 8 waves share one copy instead of 8x L2 reads.
// ---------------------------------------------------------------------------

typedef __attribute__((ext_vector_type(16))) __bf16 bf16x16;
typedef __attribute__((ext_vector_type(8)))  __bf16 bf16x8;
typedef __attribute__((ext_vector_type(8)))  float  f32x8;
typedef __attribute__((ext_vector_type(4)))  float  f32x4;

#define B_    32
#define F_    1024
#define DIN_  64
#define DOUT_ 64
#define H_    4

// ---- fragment loaders -------------------------------------------------------
// A-matrix 16x32 bf16 (ISA 7.12.2): lane L -> row m=L&15; elems 0..7 = K at
// (L>>4)*8 .. +8, elems 8..15 = K at 16+(L>>4)*8 .. +8  (two 16B chunks).
// B-matrix 32x16 bf16: lane L -> col n=L&15; same K order as A (two 16B chunks)
// -> all B operands stored N-major (transposed).

__device__ __forceinline__ bf16x16 cat8(bf16x8 lo, bf16x8 hi) {
  bf16x16 r;
#pragma unroll
  for (int i = 0; i < 8; ++i) { r[i] = lo[i]; r[8 + i] = hi[i]; }
  return r;
}

__device__ __forceinline__ bf16x16 load_a_bf16(const __bf16* base, int ld, int kofs, int lane) {
  const int m = lane & 15, g = lane >> 4;
  const __bf16* p = base + (size_t)m * ld + kofs + g * 8;
  return cat8(*(const bf16x8*)p, *(const bf16x8*)(p + 16));
}

__device__ __forceinline__ bf16x16 load_a_f32(const float* base, int ld, int kofs, int lane) {
  const int m = lane & 15, g = lane >> 4;
  const float* p = base + (size_t)m * ld + kofs + g * 8;
  f32x4 c0 = *(const f32x4*)(p);
  f32x4 c1 = *(const f32x4*)(p + 4);
  f32x4 c2 = *(const f32x4*)(p + 16);
  f32x4 c3 = *(const f32x4*)(p + 20);
  bf16x16 a;
#pragma unroll
  for (int i = 0; i < 4; ++i) {
    a[i]      = (__bf16)c0[i];
    a[4 + i]  = (__bf16)c1[i];
    a[8 + i]  = (__bf16)c2[i];
    a[12 + i] = (__bf16)c3[i];
  }
  return a;
}

__device__ __forceinline__ bf16x16 load_b_bf16(const __bf16* base, int ld, int kofs, int lane) {
  const int n = lane & 15, g = lane >> 4;
  const __bf16* p = base + (size_t)n * ld + kofs + g * 8;
  return cat8(*(const bf16x8*)p, *(const bf16x8*)(p + 16));
}

__device__ __forceinline__ f32x8 wmma32(bf16x16 a, bf16x16 b, f32x8 c) {
  return __builtin_amdgcn_wmma_f32_16x16x32_bf16(false, a, false, b, (short)0, c,
                                                 false, false);
}

// ---- CDNA5 async Global->LDS (ASYNCcnt) ------------------------------------
// Per-lane: 16 bytes memory -> LDS. VDST VGPR carries the LDS byte address
// (low 32 bits of the generic shared pointer; flat->LDS drops aperture bits).
__device__ __forceinline__ void async_ld_b128(unsigned lds_off, const void* gaddr) {
  asm volatile("global_load_async_to_lds_b128 %0, %1, off"
               :: "v"(lds_off), "v"(gaddr)
               : "memory");
}
__device__ __forceinline__ void wait_async0() {
  asm volatile("s_wait_asynccnt 0x0" ::: "memory");
}

// ---- kernel 0: weights -> bf16, transposed to N-major ----------------------
__global__ void prep_weights(const float* __restrict__ Wqkv, const float* __restrict__ Wo,
                             __bf16* __restrict__ wt, __bf16* __restrict__ wot) {
  int idx = blockIdx.x * blockDim.x + threadIdx.x;
  int stride = gridDim.x * blockDim.x;
  // wt[(h*3+kk)][o][i] = Wqkv[h][kk][i][o]   (12 x 64 x 64)
  for (int t = idx; t < 12 * 64 * 64; t += stride) {
    int hk = t >> 12, o = (t >> 6) & 63, i = t & 63;
    wt[t] = (__bf16)Wqkv[(hk << 12) + (i << 6) + o];
  }
  // wot[o][j] = Wo[j][o]   (64 x 256)
  for (int t = idx; t < 64 * 256; t += stride) {
    int o = t >> 8, j = t & 255;
    wot[t] = (__bf16)Wo[(j << 6) + o];
  }
}

// ---- kernel 1: QKV projections ---------------------------------------------
__global__ void __launch_bounds__(256, 2)
qkv_kernel(const float* __restrict__ x, const __bf16* __restrict__ wt,
           __bf16* __restrict__ qb, __bf16* __restrict__ kb, __bf16* __restrict__ vt) {
  const int lane = threadIdx.x & 31, wave = threadIdx.x >> 5;
  const int b  = blockIdx.x >> 3;                       // F_/128 = 8 tiles per b
  const int f0 = ((blockIdx.x & 7) << 7) + (wave << 4);
  const float* xrow = x + ((size_t)b * F_ + f0) * DIN_;
  bf16x16 a0 = load_a_f32(xrow, DIN_, 0, lane);
  bf16x16 a1 = load_a_f32(xrow, DIN_, 32, lane);
  const int m = lane & 15, g = lane >> 4;

  for (int h = 0; h < H_; ++h) {
    for (int kk = 0; kk < 3; ++kk) {
      const __bf16* w = wt + ((h * 3 + kk) << 12);  // 64x64, o-major
#pragma unroll
      for (int nt = 0; nt < 4; ++nt) {
        f32x8 acc = {};
        acc = wmma32(a0, load_b_bf16(w + (size_t)(nt << 4) * DIN_, DIN_, 0, lane), acc);
        acc = wmma32(a1, load_b_bf16(w + (size_t)(nt << 4) * DIN_, DIN_, 32, lane), acc);
        const int col = (nt << 4) + m;
        if (kk < 2) {
          __bf16* dst = (kk == 0 ? qb : kb) + ((size_t)(b * H_ + h) * F_ + f0) * DOUT_;
#pragma unroll
          for (int r = 0; r < 8; ++r)
            dst[(size_t)((g << 3) + r) * DOUT_ + col] = (__bf16)acc[r];
        } else {
          __bf16* dst = vt + ((size_t)(b * H_ + h) * DOUT_ + col) * F_ + f0;
#pragma unroll
          for (int r = 0; r < 8; ++r) dst[(g << 3) + r] = (__bf16)acc[r];
        }
      }
    }
  }
}

// ---- kernel 2: attention (relu(QK^T/8) @ V), async-LDS staged K/V ----------
// Block = 8 waves = one (b,h) x 128 Q rows. K/V 32-key tiles (4KB each) are
// DMA'd into LDS once per block (double-buffered), all waves consume them as
// ds_load_b128 B-fragments. relu(S) bounces C-layout -> A-layout through a
// wave-private LDS slab (LDS is in-order per wave).
__global__ void __launch_bounds__(256, 2)
attn_kernel(const __bf16* __restrict__ q, const __bf16* __restrict__ k,
            const __bf16* __restrict__ vt, __bf16* __restrict__ concat) {
  __shared__ __align__(16) __bf16 kbuf[2][32 * 64];   // 2 x 4KB  (rows g, ld=64)
  __shared__ __align__(16) __bf16 vbuf[2][64 * 32];   // 2 x 4KB  (rows d, ld=32)
  __shared__ __align__(16) __bf16 pslab[8][16 * 32];  // per-wave relu(S) slab
  const int tid  = threadIdx.x;
  const int lane = tid & 31, wave = tid >> 5;
  const int bh = blockIdx.x >> 3;
  const int f0 = ((blockIdx.x & 7) << 7) + (wave << 4);
  const int b = bh >> 2, h = bh & 3;
  const __bf16* qrow  = q + ((size_t)bh * F_ + f0) * DOUT_;
  const char*   kbase = (const char*)(k + (size_t)bh * F_ * DOUT_);
  const char*   vbase = (const char*)(vt + (size_t)bh * DOUT_ * F_);
  const int m = lane & 15, g = lane >> 4;
  const float scale = 0.125f;  // 1/sqrt(64)

  // per-thread staging slots: 16B of K (contiguous 4KB tile) + 16B of V^T
  const int vrow = tid >> 2, vcol = (tid & 3) << 4;      // V tile: 64 rows x 64B
  const unsigned klds[2] = {
      (unsigned)(uintptr_t)((const char*)&kbuf[0][0] + tid * 16),
      (unsigned)(uintptr_t)((const char*)&kbuf[1][0] + tid * 16)};
  const unsigned vlds[2] = {
      (unsigned)(uintptr_t)((const char*)&vbuf[0][0] + vrow * 64 + vcol),
      (unsigned)(uintptr_t)((const char*)&vbuf[1][0] + vrow * 64 + vcol)};
  const char* kgp = kbase + tid * 16;                    // + g0*128 per tile
  const char* vgp = vbase + (size_t)vrow * 2048 + vcol;  // + g0*2   per tile

  // prologue: stage tile 0 into buffer 0
  async_ld_b128(klds[0], kgp);
  async_ld_b128(vlds[0], vgp);

  bf16x16 aq0 = load_a_bf16(qrow, DOUT_, 0, lane);
  bf16x16 aq1 = load_a_bf16(qrow, DOUT_, 32, lane);
  f32x8 o0 = {}, o1 = {}, o2 = {}, o3 = {};
  __bf16* pl = &pslab[wave][0];

  for (int it = 0; it < F_ / 32; ++it) {
    const int buf = it & 1;
    wait_async0();       // my DMA for buf landed
    __syncthreads();     // everyone's DMA landed; everyone done reading buf^1
    if (it + 1 < F_ / 32) {
      const int g1 = (it + 1) * 32;
      async_ld_b128(klds[buf ^ 1], kgp + (size_t)g1 * 128);
      async_ld_b128(vlds[buf ^ 1], vgp + (size_t)g1 * 2);
    }
    const __bf16* kt = &kbuf[buf][0];
    const __bf16* vb = &vbuf[buf][0];
    // S tile: 16(m) x 32(g), relu'd into the wave-private LDS slab
#pragma unroll
    for (int gt = 0; gt < 32; gt += 16) {
      f32x8 s = {};
      s = wmma32(aq0, load_b_bf16(kt + gt * 64, 64, 0, lane), s);
      s = wmma32(aq1, load_b_bf16(kt + gt * 64, 64, 32, lane), s);
#pragma unroll
      for (int r = 0; r < 8; ++r) {
        float v = s[r] * scale;
        v = v > 0.f ? v : 0.f;
        pl[(((g << 3) + r) << 5) + gt + m] = (__bf16)v;  // [row][g]
      }
    }
    __builtin_amdgcn_sched_barrier(0);  // LDS is in-order per wave; fence compiler
    bf16x16 ap = load_a_bf16(pl, 32, 0, lane);
    __builtin_amdgcn_sched_barrier(0);
    // P @ V from the LDS V^T tile (d-major, ld=32): contiguous B-fragments
    o0 = wmma32(ap, load_b_bf16(vb + 0 * 16 * 32, 32, 0, lane), o0);
    o1 = wmma32(ap, load_b_bf16(vb + 1 * 16 * 32, 32, 0, lane), o1);
    o2 = wmma32(ap, load_b_bf16(vb + 2 * 16 * 32, 32, 0, lane), o2);
    o3 = wmma32(ap, load_b_bf16(vb + 3 * 16 * 32, 32, 0, lane), o3);
  }

  // write head_out directly in concat layout: [b][f][h*64 + d]  (bf16)
  __bf16* dst = concat + ((size_t)b * F_ + f0) * (H_ * DOUT_) + h * DOUT_;
  f32x8 oo[4] = {o0, o1, o2, o3};
#pragma unroll
  for (int nt = 0; nt < 4; ++nt)
#pragma unroll
    for (int r = 0; r < 8; ++r)
      dst[(size_t)((g << 3) + r) * (H_ * DOUT_) + (nt << 4) + m] = (__bf16)oo[nt][r];
}

// ---- kernel 3: output projection + bias + residual + LayerNorm -------------
__global__ void __launch_bounds__(256, 2)
proj_ln_kernel(const __bf16* __restrict__ concat, const __bf16* __restrict__ wot,
               const float* __restrict__ x, const float* __restrict__ bo,
               const float* __restrict__ gamma, const float* __restrict__ beta,
               float* __restrict__ out) {
  const int lane = threadIdx.x & 31, wave = threadIdx.x >> 5;
  const int row0 = (blockIdx.x * 8 + wave) << 4;
  const __bf16* arow = concat + (size_t)row0 * 256;
  f32x8 a0 = {}, a1 = {}, a2 = {}, a3 = {};
#pragma unroll
  for (int s = 0; s < 8; ++s) {
    bf16x16 a = load_a_bf16(arow, 256, s * 32, lane);
    a0 = wmma32(a, load_b_bf16(wot + (size_t)(0 << 4) * 256, 256, s * 32, lane), a0);
    a1 = wmma32(a, load_b_bf16(wot + (size_t)(1 << 4) * 256, 256, s * 32, lane), a1);
    a2 = wmma32(a, load_b_bf16(wot + (size_t)(2 << 4) * 256, 256, s * 32, lane), a2);
    a3 = wmma32(a, load_b_bf16(wot + (size_t)(3 << 4) * 256, 256, s * 32, lane), a3);
  }
  const int m = lane & 15, g = lane >> 4;
  const float b0 = bo[m], b1 = bo[16 + m], b2 = bo[32 + m], b3 = bo[48 + m];
  const float g0 = gamma[m], g1 = gamma[16 + m], g2 = gamma[32 + m], g3 = gamma[48 + m];
  const float e0 = beta[m], e1 = beta[16 + m], e2 = beta[32 + m], e3 = beta[48 + m];
#pragma unroll
  for (int r = 0; r < 8; ++r) {
    const int row = row0 + (g << 3) + r;
    const float* xr = x + (size_t)row * 64;
    float v0 = a0[r] + b0 + xr[m];
    float v1 = a1[r] + b1 + xr[16 + m];
    float v2 = a2[r] + b2 + xr[32 + m];
    float v3 = a3[r] + b3 + xr[48 + m];
    float s1 = v0 + v1 + v2 + v3;
    float s2 = v0 * v0 + v1 * v1 + v2 * v2 + v3 * v3;
    // lanes 0-15 / 16-31 hold different rows -> reduce within 16-lane halves
#pragma unroll
    for (int off = 1; off < 16; off <<= 1) {
      s1 += __shfl_xor(s1, off, 16);
      s2 += __shfl_xor(s2, off, 16);
    }
    const float mean = s1 * (1.f / 64.f);
    const float var = s2 * (1.f / 64.f) - mean * mean;
    const float rinv = rsqrtf(var + 1e-5f);
    float* orow = out + (size_t)row * 64;
    orow[m]      = (v0 - mean) * rinv * g0 + e0;
    orow[16 + m] = (v1 - mean) * rinv * g1 + e1;
    orow[32 + m] = (v2 - mean) * rinv * g2 + e2;
    orow[48 + m] = (v3 - mean) * rinv * g3 + e3;
  }
}

// ---- host ------------------------------------------------------------------
extern "C" void kernel_launch(void* const* d_in, const int* in_sizes, int n_in,
                              void* d_out, int out_size, void* d_ws, size_t ws_size,
                              hipStream_t stream) {
  const float* x     = (const float*)d_in[0];  // (B,F,64)
  const float* Wqkv  = (const float*)d_in[1];  // (H,3,64,64)
  const float* Wo    = (const float*)d_in[2];  // (256,64)
  const float* bo    = (const float*)d_in[3];
  const float* gamma = (const float*)d_in[4];
  const float* beta  = (const float*)d_in[5];
  float* out = (float*)d_out;

  char* ws = (char*)d_ws;
  __bf16* qb  = (__bf16*)(ws);                                 // 16 MB
  __bf16* kb  = (__bf16*)(ws + ((size_t)16 << 20));            // 16 MB
  __bf16* vt  = (__bf16*)(ws + ((size_t)32 << 20));            // 16 MB (V^T)
  __bf16* cc  = (__bf16*)(ws + ((size_t)48 << 20));            // 16 MB (concat)
  __bf16* wt  = (__bf16*)(ws + ((size_t)64 << 20));            // 96 KB (Wqkv^T)
  __bf16* wot = (__bf16*)(ws + ((size_t)64 << 20) + (128 << 10));  // 32 KB (Wo^T)

  hipLaunchKernelGGL(prep_weights, dim3(64), dim3(256), 0, stream, Wqkv, Wo, wt, wot);
  hipLaunchKernelGGL(qkv_kernel, dim3(B_ * (F_ / 128)), dim3(256), 0, stream,
                     x, wt, qb, kb, vt);
  hipLaunchKernelGGL(attn_kernel, dim3(B_ * H_ * (F_ / 128)), dim3(256), 0, stream,
                     qb, kb, vt, cc);
  hipLaunchKernelGGL(proj_ln_kernel, dim3((B_ * F_) / 128), dim3(256), 0, stream,
                     cc, wot, x, bo, gamma, beta, out);
}
